// MNIST_FCN_1640677507485
// MI455X (gfx1250) — compile-verified, matrix operand order checked by reading, not run on hardware
//
#include <hip/hip_runtime.h>

// ---------------------------------------------------------------------------
// MNIST_FCN KM fixed-point solver for MI455X (gfx1250, wave32, WMMA).
// GEMMs use split-bf16 (hi+lo) with V_WMMA_F32_16X16X32_BF16, 3 products per
// K-step for fp32-class accuracy at ~2.7x the fp32-WMMA rate.
// Double-buffered LDS staging + register prefetch hides global latency.
// ---------------------------------------------------------------------------

typedef __attribute__((ext_vector_type(16))) __bf16 v16bf;
typedef __attribute__((ext_vector_type(8)))  __bf16 v8bf;
typedef __attribute__((ext_vector_type(4)))  __bf16 v4bf;
typedef __attribute__((ext_vector_type(8)))  float  v8f;

union Frag16 { v16bf v; v8bf h[2]; };

struct Ctrl { int done; int final_sel; };

#define TILE_M 128
#define TILE_N 128
#define TILE_K 32
#define LDSS   40          // bf16 elems per LDS row (32 data + 8 pad) = 80B
#define NITER  40          // >> ~15 iters needed at contraction ~0.27
#define EPS2   1e-10f      // (1e-5)^2 on squared row residual

__device__ __forceinline__ void split4(const float4& v, v4bf& hi, v4bf& lo)
{
    hi[0] = (__bf16)v.x; lo[0] = (__bf16)(v.x - (float)hi[0]);
    hi[1] = (__bf16)v.y; lo[1] = (__bf16)(v.y - (float)hi[1]);
    hi[2] = (__bf16)v.z; lo[2] = (__bf16)(v.z - (float)hi[2]);
    hi[3] = (__bf16)v.w; lo[3] = (__bf16)(v.w - (float)hi[3]);
}

// ---------------------------------------------------------------------------
// MODE 0: Out = relu(A@W^T + bias)                     (h from d, Wd, bd)
// MODE 1: Out = 0.9 * (A@W^T)                          (y from h, Wy)
// MODE 2: t = relu(0.1*(A@W^T) + Y); Out = 0.09*Uold + 0.9*t; res2 += row sq
// KFULL : K is a multiple of TILE_K -> no edge guards in the hot loop
// ---------------------------------------------------------------------------
template <int MODE, bool KFULL>
__global__ __launch_bounds__(256)
void gemm_kernel(const float* __restrict__ A, int lda,
                 const float* __restrict__ W, int ldw,
                 const float* __restrict__ bias,
                 const float* __restrict__ Y,
                 const float* __restrict__ Uold,
                 float* __restrict__ Out,
                 float* __restrict__ res2,
                 const Ctrl* __restrict__ ctrl,
                 int M, int N, int K)
{
    if (MODE == 2 && ctrl->done) return;   // gated iteration (uniform exit)

    // [buf][hi/lo][row][k] : 2 x 2 x 128 x 40 x 2B = 40 KB each, 80 KB total
    __shared__ __align__(16) __bf16 As[2][2][TILE_M][LDSS];
    __shared__ __align__(16) __bf16 Ws[2][2][TILE_N][LDSS];

    const int tid   = threadIdx.x;
    const int wid   = tid >> 5;       // 8 waves
    const int lane  = tid & 31;
    const int wm    = wid & 3;        // 4 M-bands of 32 rows
    const int wn    = wid >> 2;       // 2 N-bands of 64 cols
    const int lhalf = lane >> 4;      // 0/1 : K-half selector (ISA layout)
    const int lrow  = lane & 15;      // row (A) / col (B) within 16
    const int blkM  = blockIdx.x * TILE_M;
    const int blkN  = blockIdx.y * TILE_N;

    v8f acc[2][4];
#pragma unroll
    for (int i = 0; i < 2; ++i)
#pragma unroll
        for (int j = 0; j < 4; ++j)
#pragma unroll
            for (int e = 0; e < 8; ++e) acc[i][j][e] = 0.0f;

    // staging decomposition: 256 threads -> 32 rows x 8 float4-cols per pass
    const int sr = tid >> 3;          // 0..31
    const int kq = (tid & 7) * 4;     // float4 column within K-tile

    float4 aReg[4], wReg[4];          // register-held prefetch of one K-tile

    auto loadTile = [&](int k0) {
#pragma unroll
        for (int rr = 0; rr < 4; ++rr) {
            const int row = rr * 32 + sr;
            if (KFULL || (k0 + kq + 4) <= K) {
                aReg[rr] = *(const float4*)(A + (size_t)(blkM + row) * lda + k0 + kq);
                wReg[rr] = *(const float4*)(W + (size_t)(blkN + row) * ldw + k0 + kq);
            } else {
                aReg[rr] = float4{0.f, 0.f, 0.f, 0.f};
                wReg[rr] = float4{0.f, 0.f, 0.f, 0.f};
            }
        }
    };
    auto storeTile = [&](int buf) {
#pragma unroll
        for (int rr = 0; rr < 4; ++rr) {
            const int row = rr * 32 + sr;
            v4bf ah, al, wh, wl;
            split4(aReg[rr], ah, al);
            split4(wReg[rr], wh, wl);
            *(v4bf*)&As[buf][0][row][kq] = ah;
            *(v4bf*)&As[buf][1][row][kq] = al;
            *(v4bf*)&Ws[buf][0][row][kq] = wh;
            *(v4bf*)&Ws[buf][1][row][kq] = wl;
        }
    };

    const int ksteps = (K + TILE_K - 1) / TILE_K;

    loadTile(0);
    storeTile(0);

    int buf = 0;
    for (int ks = 0; ks < ksteps; ++ks) {
        __syncthreads();                       // LDS[buf] ready for all waves
        const bool more = (ks + 1) < ksteps;
        if (more) loadTile((ks + 1) * TILE_K); // global fetch overlaps compute

        // ---- fragments per ISA 16-bit A/B layout ---------------------------
        // lane L (L<16): K 0..7 & 16..23 of row L; lane L+16: K 8..15 & 24..31
        Frag16 ah[2], al[2], bh[4], bl[4];
#pragma unroll
        for (int i = 0; i < 2; ++i) {
            const int r = wm * 32 + i * 16 + lrow;
            ah[i].h[0] = *(const v8bf*)&As[buf][0][r][lhalf * 8];
            ah[i].h[1] = *(const v8bf*)&As[buf][0][r][16 + lhalf * 8];
            al[i].h[0] = *(const v8bf*)&As[buf][1][r][lhalf * 8];
            al[i].h[1] = *(const v8bf*)&As[buf][1][r][16 + lhalf * 8];
        }
#pragma unroll
        for (int j = 0; j < 4; ++j) {
            const int c = wn * 64 + j * 16 + lrow;
            bh[j].h[0] = *(const v8bf*)&Ws[buf][0][c][lhalf * 8];
            bh[j].h[1] = *(const v8bf*)&Ws[buf][0][c][16 + lhalf * 8];
            bl[j].h[0] = *(const v8bf*)&Ws[buf][1][c][lhalf * 8];
            bl[j].h[1] = *(const v8bf*)&Ws[buf][1][c][16 + lhalf * 8];
        }
        // ---- split-bf16 3-product accumulation (fp32-class precision) ------
#pragma unroll
        for (int i = 0; i < 2; ++i)
#pragma unroll
            for (int j = 0; j < 4; ++j) {
                acc[i][j] = __builtin_amdgcn_wmma_f32_16x16x32_bf16(
                    false, ah[i].v, false, bh[j].v, (short)0, acc[i][j], false, false);
                acc[i][j] = __builtin_amdgcn_wmma_f32_16x16x32_bf16(
                    false, ah[i].v, false, bl[j].v, (short)0, acc[i][j], false, false);
                acc[i][j] = __builtin_amdgcn_wmma_f32_16x16x32_bf16(
                    false, al[i].v, false, bh[j].v, (short)0, acc[i][j], false, false);
            }

        if (more) storeTile(buf ^ 1);          // fill the other buffer
        buf ^= 1;
    }

    // ---- epilogue: C/D layout: lane -> (m = lhalf*8 + g, n = lrow) ---------
    float rs[2][8];
    if (MODE == 2) {
#pragma unroll
        for (int i = 0; i < 2; ++i)
#pragma unroll
            for (int g = 0; g < 8; ++g) rs[i][g] = 0.0f;
    }
#pragma unroll
    for (int i = 0; i < 2; ++i) {
        const int rbase = blkM + wm * 32 + i * 16 + lhalf * 8;
#pragma unroll
        for (int j = 0; j < 4; ++j) {
            const int c = blkN + wn * 64 + j * 16 + lrow;
            const float b = (MODE == 0) ? bias[c] : 0.0f;
#pragma unroll
            for (int g = 0; g < 8; ++g) {
                const size_t idx = (size_t)(rbase + g) * N + c;
                float v = acc[i][j][g];
                if (MODE == 0) {
                    v += b;
                    Out[idx] = v > 0.0f ? v : 0.0f;
                } else if (MODE == 1) {
                    Out[idx] = 0.9f * v;
                } else {
                    const float uo = Uold[idx];
                    float t = 0.1f * v + Y[idx];
                    t = t > 0.0f ? t : 0.0f;
                    const float un = 0.09f * uo + 0.9f * t;
                    Out[idx] = un;
                    const float dd = un - uo;
                    rs[i][g] += dd * dd;
                }
            }
        }
    }
    if (MODE == 2) {
        // reduce squared diffs across the 16 lanes holding each row
#pragma unroll
        for (int i = 0; i < 2; ++i)
#pragma unroll
            for (int g = 0; g < 8; ++g) {
                float v = rs[i][g];
                v += __shfl_xor(v, 1, 16);
                v += __shfl_xor(v, 2, 16);
                v += __shfl_xor(v, 4, 16);
                v += __shfl_xor(v, 8, 16);
                if (lrow == 0) {
                    const int r = blkM + wm * 32 + i * 16 + lhalf * 8 + g;
                    atomicAdd(&res2[r], v);
                }
            }
    }
}

// ---------------------------------------------------------------------------
__global__ void init_kernel(float* __restrict__ u0, float* __restrict__ res2,
                            int* __restrict__ nc, Ctrl* __restrict__ ctrl,
                            size_t total, int ncols)
{
    const size_t stride = (size_t)gridDim.x * blockDim.x;
    for (size_t i = (size_t)blockIdx.x * blockDim.x + threadIdx.x; i < total; i += stride) {
        const int col = (int)(i % (size_t)ncols);
        u0[i] = (col < 10) ? 0.1f : 0.0f;
    }
    const size_t tidg = (size_t)blockIdx.x * blockDim.x + threadIdx.x;
    if (tidg < 8192) { res2[tidg] = 0.0f; nc[tidg] = 1; }
    if (tidg == 0) { ctrl->done = 0; ctrl->final_sel = 0; }
}

// sticky per-row convergence flags + global done flag; resets res2 for next it
__global__ void check_kernel(Ctrl* __restrict__ ctrl, float* __restrict__ res2,
                             int* __restrict__ nc, int next_sel, int rows)
{
    if (ctrl->done) return;
    __shared__ int cnt;
    if (threadIdx.x == 0) cnt = 0;
    __syncthreads();
    int local = 0;
    for (int r = threadIdx.x; r < rows; r += blockDim.x) {
        int f = nc[r];
        if (f && res2[r] <= EPS2) f = 0;   // converged rows stay converged
        nc[r] = f;
        local += f;
        res2[r] = 0.0f;
    }
    atomicAdd(&cnt, local);
    __syncthreads();
    if (threadIdx.x == 0) {
        ctrl->final_sel = next_sel;        // buffer that now holds current u
        if (cnt == 0) ctrl->done = 1;
    }
}

__global__ void finalize_kernel(const Ctrl* __restrict__ ctrl,
                                const float* __restrict__ uA,
                                float* __restrict__ out, size_t total)
{
    if (ctrl->final_sel != 0) return;      // final u already in d_out buffer
    const size_t stride = (size_t)gridDim.x * blockDim.x;
    for (size_t i = (size_t)blockIdx.x * blockDim.x + threadIdx.x; i < total; i += stride)
        out[i] = uA[i];
}

// ---------------------------------------------------------------------------
extern "C" void kernel_launch(void* const* d_in, const int* in_sizes, int n_in,
                              void* d_out, int out_size, void* d_ws, size_t ws_size,
                              hipStream_t stream)
{
    const float* d  = (const float*)d_in[0];   // [8192, 784]
    const float* Wd = (const float*)d_in[1];   // [2048, 784]
    const float* bd = (const float*)d_in[2];   // [2048]
    const float* Wu = (const float*)d_in[3];   // [2048, 2048]
    const float* Wy = (const float*)d_in[4];   // [2048, 2048]
    float* out = (float*)d_out;                // [8192, 2048]

    const int M = 8192, N = 2048, Kd = 784, Kn = 2048;
    const size_t total = (size_t)M * N;

    // workspace layout (needs ~128.1 MB)
    char* ws = (char*)d_ws;
    Ctrl*  ctrl = (Ctrl*)ws;                         // 256 B
    float* res2 = (float*)(ws + 256);                // 32 KB
    int*   nc   = (int*)(ws + 256 + 32768);          // 32 KB
    float* bufA = (float*)(ws + 256 + 65536);        // 64 MB: h, then u ping
    float* y    = (float*)((char*)bufA + total * 4); // 64 MB
    float* uB   = out;                               // u pong lives in d_out

    dim3 grid(M / TILE_M, N / TILE_N), block(256);

    // h = relu(d @ Wd^T + bd)   (stored in bufA); K=784 -> guarded edge path
    gemm_kernel<0, false><<<grid, block, 0, stream>>>(d, Kd, Wd, Kd, bd,
                                                      nullptr, nullptr, bufA,
                                                      nullptr, ctrl, M, N, Kd);
    // y = 0.9 * (h @ Wy^T); K=2048 -> unguarded hot loop
    gemm_kernel<1, true><<<grid, block, 0, stream>>>(bufA, Kn, Wy, Kn, nullptr,
                                                     nullptr, nullptr, y,
                                                     nullptr, ctrl, M, N, Kn);
    // u0 (overwrites h region) + flags/ctrl
    init_kernel<<<8192, 256, 0, stream>>>(bufA, res2, nc, ctrl, total, N);

    // gated KM iterations (no-ops once ctrl->done is set)
    for (int i = 0; i < NITER; ++i) {
        const float* ucur = (i & 1) ? uB : bufA;
        float*       unxt = (i & 1) ? bufA : uB;
        gemm_kernel<2, true><<<grid, block, 0, stream>>>(ucur, Kn, Wu, Kn, nullptr,
                                                         y, ucur, unxt,
                                                         res2, ctrl, M, N, Kn);
        check_kernel<<<1, 256, 0, stream>>>(ctrl, res2, nc, (i + 1) & 1, M);
    }

    // if the final buffer is bufA, copy it into d_out
    finalize_kernel<<<8192, 256, 0, stream>>>(ctrl, bufA, out, total);
}
